// SparseCausalSelfAttention_24524263260729
// MI455X (gfx1250) — matrix operand order repaired and checked by reading, not compiled, not run on hardware
//
#include <hip/hip_runtime.h>
#include <hip/hip_bf16.h>
#include <stddef.h>
#include <stdint.h>

// ---------------- problem constants ----------------
#define Bb 2
#define Tt 512
#define Cc 1024
#define Ee 8
#define TOPK 2
#define NHh 8
#define HSs 128
#define AHh 1024
#define BT (Bb*Tt)          // 1024 tokens

// ---------------- workspace layout (bytes) ----------------
#define OFF_COUNTS   0            //  8 ints
#define OFF_LISTS    256          //  E*BT ints   = 32768
#define OFF_GATES    33024        //  BT*2 f32    = 8192
#define OFF_PART     41216        //  128*8 f32   = 4096
#define OFF_KBF      45312        //  BT*AH bf16  = 2 MB
#define OFF_VBF      2142464     //  BT*AH bf16  = 2 MB
#define OFF_QBF      4239616     //  BT*2*AH bf16= 4 MB
#define OFF_YWBF     8433920     //  BT*2*AH bf16= 4 MB
#define OFF_YPROJ    12628224    //  BT*2*C f32  = 8 MB

typedef __attribute__((ext_vector_type(16))) __bf16 bf16x16;
typedef __attribute__((ext_vector_type(8)))  __bf16 bf16x8;
typedef __attribute__((ext_vector_type(4)))  __bf16 bf16x4;
typedef __attribute__((ext_vector_type(8)))  float  floatx8;

// ---- CDNA5 async copy: global -> LDS, 16 bytes per lane (ASYNCcnt) ----
__device__ __forceinline__ void async_copy16(void* lds_dst, const void* gsrc) {
  unsigned           l = (unsigned)(uintptr_t)lds_dst;          // addr[31:0] = LDS offset
  unsigned long long g = (unsigned long long)(uintptr_t)gsrc;
  asm volatile("global_load_async_to_lds_b128 %0, %1, off" :: "v"(l), "v"(g) : "memory");
}
__device__ __forceinline__ void wait_async0() {
  asm volatile("s_wait_asynccnt 0" ::: "memory");
}

// A operand: 16x32 bf16 tile from row-major LDS [m][k], row stride ldk.
__device__ __forceinline__ bf16x16 lds_load_A(const __bf16* base, int ldk, int lane) {
  int m  = lane & 15;
  int kb = (lane & 16) ? 8 : 0;
  const __bf16* p = base + m * ldk + kb;
  union { bf16x16 v; bf16x8 h[2]; } u;
  u.h[0] = *(const bf16x8*)(p);
  u.h[1] = *(const bf16x8*)(p + 16);
  return u.v;
}
// B operand: 32x16 bf16 tile from N-major LDS [n][k], row stride ldk.
__device__ __forceinline__ bf16x16 lds_load_B(const __bf16* base, int ldk, int lane) {
  int n  = lane & 15;
  int kb = (lane & 16) ? 8 : 0;
  const __bf16* p = base + n * ldk + kb;
  union { bf16x16 v; bf16x8 h[2]; } u;
  u.h[0] = *(const bf16x8*)(p);
  u.h[1] = *(const bf16x8*)(p + 16);
  return u.v;
}
#define WMMA_BF16(a,b,c) \
  __builtin_amdgcn_wmma_f32_16x16x32_bf16(false,(a),false,(b),(short)0,(c),false,false)

__device__ __forceinline__ bf16x4 cvt4(float4 f) {
  bf16x4 h; h[0] = (__bf16)f.x; h[1] = (__bf16)f.y; h[2] = (__bf16)f.z; h[3] = (__bf16)f.w;
  return h;
}

// =====================================================================
// Kernel 1: gating. 8 waves/block, one token per wave.
// =====================================================================
__global__ void gate_kernel(const float* __restrict__ x, const float* __restrict__ Wg,
                            float* __restrict__ gates, int* __restrict__ counts,
                            int* __restrict__ lists, float* __restrict__ partials) {
  int lane = threadIdx.x & 31, wave = threadIdx.x >> 5;
  int tok  = blockIdx.x * 8 + wave;
  float acc[Ee];
#pragma unroll
  for (int e = 0; e < Ee; ++e) acc[e] = 0.f;
  for (int c = lane; c < Cc; c += 32) {
    float xv = x[(size_t)tok * Cc + c];
    const float* w = Wg + (size_t)c * Ee;
#pragma unroll
    for (int e = 0; e < Ee; ++e) acc[e] += xv * w[e];
  }
#pragma unroll
  for (int off = 16; off > 0; off >>= 1)
#pragma unroll
    for (int e = 0; e < Ee; ++e) acc[e] += __shfl_down(acc[e], off, 32);

  __shared__ float sp[8][Ee];
  if (lane == 0) {
    int i0 = 0;
#pragma unroll
    for (int e = 1; e < Ee; ++e) if (acc[e] > acc[i0]) i0 = e;
    int i1 = (i0 == 0) ? 1 : 0;
#pragma unroll
    for (int e = 0; e < Ee; ++e) if (e != i0 && acc[e] > acc[i1]) i1 = e;
    float g0 = 1.f / (1.f + __expf(acc[i1] - acc[i0]));   // softmax of top-2
    gates[tok * 2 + 0] = g0;
    gates[tok * 2 + 1] = 1.f - g0;
    int p0 = atomicAdd(&counts[i0], 1); lists[i0 * BT + p0] = tok * 2 + 0;
    int p1 = atomicAdd(&counts[i1], 1); lists[i1 * BT + p1] = tok * 2 + 1;
    float mx = acc[0];
#pragma unroll
    for (int e = 1; e < Ee; ++e) mx = fmaxf(mx, acc[e]);
    float se = 0.f, pr[Ee];
#pragma unroll
    for (int e = 0; e < Ee; ++e) { pr[e] = __expf(acc[e] - mx); se += pr[e]; }
#pragma unroll
    for (int e = 0; e < Ee; ++e) sp[wave][e] = pr[e] / se;
  }
  __syncthreads();
  if (threadIdx.x == 0) {          // fixed-order partial sum -> deterministic
#pragma unroll
    for (int e = 0; e < Ee; ++e) {
      float s = 0.f;
      for (int w = 0; w < 8; ++w) s += sp[w][e];
      partials[blockIdx.x * Ee + e] = s;
    }
  }
}

// =====================================================================
// Kernel 2: K/V projection (z=0 -> K, z=1 -> V). 64x64 tile, K-step 64.
// =====================================================================
__global__ void kv_kernel(const float* __restrict__ x,
                          const float* __restrict__ Wk, const float* __restrict__ bk,
                          const float* __restrict__ Wv, const float* __restrict__ bv,
                          __bf16* __restrict__ kbf, __bf16* __restrict__ vbf) {
  const float* W    = blockIdx.z ? Wv : Wk;
  const float* bias = blockIdx.z ? bv : bk;
  __bf16*      out  = blockIdx.z ? vbf : kbf;
  __shared__ __bf16 sA[64 * 64];     // [m][k]  8 KB
  __shared__ __bf16 sB[64 * 64];     // [n][k]  8 KB
  int t = threadIdx.x, lane = t & 31, w = t >> 5;
  int m0 = blockIdx.y * 64, n0 = blockIdx.x * 64;
  int msub = w & 3, npair = w >> 2;
  floatx8 acc0 = {}, acc1 = {};
  for (int kk = 0; kk < Cc; kk += 64) {
#pragma unroll
    for (int i = 0; i < 4; ++i) {                 // A: 1024 float4 chunks
      int idx = t + i * 256;
      int kr4 = idx & 15, mr = idx >> 4;
      float4 f = *(const float4*)(x + (size_t)(m0 + mr) * Cc + kk + kr4 * 4);
      *(bf16x4*)(sA + mr * 64 + kr4 * 4) = cvt4(f);
    }
#pragma unroll
    for (int i = 0; i < 4; ++i) {                 // B: transpose into [n][k]
      int idx = t + i * 256;
      int nc4 = idx & 15, kr = idx >> 4;
      float4 f = *(const float4*)(W + (size_t)(kk + kr) * AHh + n0 + nc4 * 4);
      sB[(nc4 * 4 + 0) * 64 + kr] = (__bf16)f.x;
      sB[(nc4 * 4 + 1) * 64 + kr] = (__bf16)f.y;
      sB[(nc4 * 4 + 2) * 64 + kr] = (__bf16)f.z;
      sB[(nc4 * 4 + 3) * 64 + kr] = (__bf16)f.w;
    }
    if (kk + 64 < Cc) __builtin_prefetch(W + (size_t)(kk + 64) * AHh + n0, 0, 1);
    __syncthreads();
#pragma unroll
    for (int ks = 0; ks < 64; ks += 32) {
      bf16x16 a  = lds_load_A(sA + msub * 16 * 64 + ks, 64, lane);
      bf16x16 b0 = lds_load_B(sB + (npair * 32) * 64 + ks, 64, lane);
      bf16x16 b1 = lds_load_B(sB + (npair * 32 + 16) * 64 + ks, 64, lane);
      acc0 = WMMA_BF16(a, b0, acc0);
      acc1 = WMMA_BF16(a, b1, acc1);
    }
    __syncthreads();
  }
  int n = lane & 15, mh = (lane & 16) ? 8 : 0;
#pragma unroll
  for (int r = 0; r < 8; ++r) {
    int m  = m0 + msub * 16 + mh + r;
    int c0 = n0 + npair * 32 + n;
    out[(size_t)m * AHh + c0]      = (__bf16)(acc0[r] + bias[c0]);
    out[(size_t)m * AHh + c0 + 16] = (__bf16)(acc1[r] + bias[c0 + 16]);
  }
}

// =====================================================================
// Kernel 3: grouped Q projection: rows gathered from expert token lists.
// =====================================================================
__global__ void qproj_kernel(const float* __restrict__ x, const float* __restrict__ W_in,
                             const int* __restrict__ counts, const int* __restrict__ lists,
                             __bf16* __restrict__ qbf) {
  int e   = blockIdx.z;
  int n_e = __builtin_amdgcn_readfirstlane(counts[e]);
  int m0  = blockIdx.y * 64;
  if (m0 >= n_e) return;
  __shared__ __bf16 sA[64 * 64];
  __shared__ __bf16 sB[64 * 64];
  __shared__ int    srow[64];
  int t = threadIdx.x, lane = t & 31, w = t >> 5;
  if (t < 64) srow[t] = (m0 + t < n_e) ? lists[e * BT + m0 + t] : -1;
  __syncthreads();
  int n0 = blockIdx.x * 64;
  const float* W = W_in + (size_t)e * Cc * AHh;
  int msub = w & 3, npair = w >> 2;
  floatx8 acc0 = {}, acc1 = {};
  for (int kk = 0; kk < Cc; kk += 64) {
#pragma unroll
    for (int i = 0; i < 4; ++i) {                 // A: gathered token rows
      int idx = t + i * 256;
      int kr4 = idx & 15, mr = idx >> 4;
      int en  = srow[mr];
      int row = (en >= 0) ? (en >> 1) : 0;        // dummy row for invalid (not stored)
      float4 f = *(const float4*)(x + (size_t)row * Cc + kk + kr4 * 4);
      *(bf16x4*)(sA + mr * 64 + kr4 * 4) = cvt4(f);
    }
#pragma unroll
    for (int i = 0; i < 4; ++i) {
      int idx = t + i * 256;
      int nc4 = idx & 15, kr = idx >> 4;
      float4 f = *(const float4*)(W + (size_t)(kk + kr) * AHh + n0 + nc4 * 4);
      sB[(nc4 * 4 + 0) * 64 + kr] = (__bf16)f.x;
      sB[(nc4 * 4 + 1) * 64 + kr] = (__bf16)f.y;
      sB[(nc4 * 4 + 2) * 64 + kr] = (__bf16)f.z;
      sB[(nc4 * 4 + 3) * 64 + kr] = (__bf16)f.w;
    }
    if (kk + 64 < Cc) __builtin_prefetch(W + (size_t)(kk + 64) * AHh + n0, 0, 1);
    __syncthreads();
#pragma unroll
    for (int ks = 0; ks < 64; ks += 32) {
      bf16x16 a  = lds_load_A(sA + msub * 16 * 64 + ks, 64, lane);
      bf16x16 b0 = lds_load_B(sB + (npair * 32) * 64 + ks, 64, lane);
      bf16x16 b1 = lds_load_B(sB + (npair * 32 + 16) * 64 + ks, 64, lane);
      acc0 = WMMA_BF16(a, b0, acc0);
      acc1 = WMMA_BF16(a, b1, acc1);
    }
    __syncthreads();
  }
  int n = lane & 15, mh = (lane & 16) ? 8 : 0;
#pragma unroll
  for (int r = 0; r < 8; ++r) {
    int row = msub * 16 + mh + r;
    int en  = srow[row];
    if (en >= 0) {
      int c0 = n0 + npair * 32 + n;
      qbf[(size_t)en * AHh + c0]      = (__bf16)acc0[r];
      qbf[(size_t)en * AHh + c0 + 16] = (__bf16)acc1[r];
    }
  }
}

// =====================================================================
// Kernel 4: stick-breaking flash attention per (b,slot,head, 64-query tile)
// j tiles of 32 keys, descending from the diagonal. Q/K staged by async DMA.
// =====================================================================
__global__ void attn_kernel(const __bf16* __restrict__ qbf, const __bf16* __restrict__ kbf,
                            const __bf16* __restrict__ vbf, const float* __restrict__ gates,
                            __bf16* __restrict__ ywbf) {
  int it = blockIdx.x;                 // query tile (64 rows)
  int h  = blockIdx.y;                 // head
  int bz = blockIdx.z;                 // b*2+slot
  int b = bz >> 1, slot = bz & 1;
  int i0 = it * 64;
  __shared__ __bf16 sQ[64 * 128];      // 16 KB
  __shared__ __bf16 sK[32 * 128];      //  8 KB
  __shared__ __bf16 sVT[128 * 32];     //  8 KB  [hs][j]
  __shared__ float  sS[64 * 32];       //  8 KB
  __shared__ __bf16 sAtt[64 * 32];     //  4 KB
  int t = threadIdx.x, lane = t & 31, w = t >> 5;
  int msub = w & 3, nsub = w >> 2;     // S phase: 4m x 2n subtiles
  int hgrp = w >> 2;                   // y phase: 4m x (2 groups of 4 hs subtiles)

#pragma unroll
  for (int i = 0; i < 4; ++i) {        // async Q tile: 1024 x 16B chunks
    int c = t + i * 256;
    int row = c >> 4, cc = c & 15;
    int tok = b * Tt + i0 + row;
    async_copy16(sQ + row * 128 + cc * 8,
                 qbf + (size_t)(tok * 2 + slot) * AHh + h * HSs + cc * 8);
  }
  float carry = 0.f;
  floatx8 y0 = {}, y1 = {}, y2 = {}, y3 = {};
  for (int jt = it * 2 + 1; jt >= 0; --jt) {
    int j0 = jt * 32;
#pragma unroll
    for (int i = 0; i < 2; ++i) {      // async K tile: 512 x 16B chunks
      int c = t + i * 256;
      int row = c >> 4, cc = c & 15;
      int tok = b * Tt + j0 + row;
      async_copy16(sK + row * 128 + cc * 8,
                   kbf + (size_t)tok * AHh + h * HSs + cc * 8);
    }
#pragma unroll
    for (int i = 0; i < 2; ++i) {      // V tile, transposed into [hs][j]
      int idx = t + i * 256;
      int hh8 = idx & 15, jj = idx >> 4;
      bf16x8 v = *(const bf16x8*)(vbf + (size_t)(b * Tt + j0 + jj) * AHh + h * HSs + hh8 * 8);
#pragma unroll
      for (int q = 0; q < 8; ++q) sVT[(hh8 * 8 + q) * 32 + jj] = v[q];
    }
    wait_async0();
    __syncthreads();
    // ---- S = Q K^T * scale
    floatx8 s0 = {};
#pragma unroll
    for (int kk = 0; kk < HSs; kk += 32) {
      bf16x16 a  = lds_load_A(sQ + msub * 16 * 128 + kk, 128, lane);
      bf16x16 bm = lds_load_B(sK + (nsub * 16) * 128 + kk, 128, lane);
      s0 = WMMA_BF16(a, bm, s0);
    }
    {
      const float scale = 0.08838834764831845f;  // 1/sqrt(128)
      int n = lane & 15, mh = (lane & 16) ? 8 : 0;
#pragma unroll
      for (int r = 0; r < 8; ++r)
        sS[(msub * 16 + mh + r) * 32 + nsub * 16 + n] = s0[r] * scale;
    }
    __syncthreads();
    // ---- per-row stick-breaking suffix scan (descending j)
    if (t < 64) {
      int ig = i0 + t;
      float suf = 0.f;
      for (int jj = 31; jj >= 0; --jj) {
        int jg = j0 + jj;
        float s  = sS[t * 32 + jj];
        float lz = fminf(s, 0.f) - log1pf(__expf(-fabsf(s)));   // log sigmoid(s)
        float att = (jg <= ig) ? __expf(lz + carry + suf) : 0.f;
        sAtt[t * 32 + jj] = (__bf16)att;
        if (jg <= ig) suf += lz - s;                            // log(1-sigmoid(s))
      }
      carry += suf;
    }
    __syncthreads();
    // ---- y += att * V   (kdim = 32)
    {
      bf16x16 a  = lds_load_A(sAtt + msub * 16 * 32, 32, lane);
      bf16x16 b0 = lds_load_B(sVT + (hgrp * 64 + 0)  * 32, 32, lane);
      bf16x16 b1 = lds_load_B(sVT + (hgrp * 64 + 16) * 32, 32, lane);
      bf16x16 b2 = lds_load_B(sVT + (hgrp * 64 + 32) * 32, 32, lane);
      bf16x16 b3 = lds_load_B(sVT + (hgrp * 64 + 48) * 32, 32, lane);
      y0 = WMMA_BF16(a, b0, y0);
      y1 = WMMA_BF16(a, b1, y1);
      y2 = WMMA_BF16(a, b2, y2);
      y3 = WMMA_BF16(a, b3, y3);
    }
    __syncthreads();
  }
  // ---- gate and store yw (bf16) for the out projection
  int n = lane & 15, mh = (lane & 16) ? 8 : 0;
#pragma unroll
  for (int r = 0; r < 8; ++r) {
    int m   = msub * 16 + mh + r;
    int tok = b * Tt + i0 + m;
    float g = gates[tok * 2 + slot];
    size_t base = (size_t)(tok * 2 + slot) * AHh + h * HSs + hgrp * 64 + n;
    ywbf[base + 0]  = (__bf16)(y0[r] * g);
    ywbf[base + 16] = (__bf16)(y1[r] * g);
    ywbf[base + 32] = (__bf16)(y2[r] * g);
    ywbf[base + 48] = (__bf16)(y3[r] * g);
  }
}

// =====================================================================
// Kernel 5: grouped output projection yw @ W_out[e] -> yproj (f32)
// A tile (already bf16) staged by async DMA.
// =====================================================================
__global__ void oproj_kernel(const __bf16* __restrict__ ywbf, const float* __restrict__ W_out,
                             const int* __restrict__ counts, const int* __restrict__ lists,
                             float* __restrict__ yproj) {
  int e   = blockIdx.z;
  int n_e = __builtin_amdgcn_readfirstlane(counts[e]);
  int m0  = blockIdx.y * 64;
  if (m0 >= n_e) return;
  __shared__ __bf16 sA[64 * 64];
  __shared__ __bf16 sB[64 * 64];
  __shared__ int    srow[64];
  int t = threadIdx.x, lane = t & 31, w = t >> 5;
  if (t < 64) srow[t] = (m0 + t < n_e) ? lists[e * BT + m0 + t] : -1;
  __syncthreads();
  int en0 = srow[0];                                  // always valid (m0 < n_e)
  int n0 = blockIdx.x * 64;
  const float* W = W_out + (size_t)e * AHh * Cc;
  int msub = w & 3, npair = w >> 2;
  floatx8 acc0 = {}, acc1 = {};
  for (int kk = 0; kk < AHh; kk += 64) {
#pragma unroll
    for (int i = 0; i < 2; ++i) {                     // async A: 512 x 16B chunks
      int idx = t + i * 256;
      int cc = idx & 7, mr = idx >> 3;
      int en = srow[mr]; if (en < 0) en = en0;        // dummy row (not stored)
      async_copy16(sA + mr * 64 + cc * 8,
                   ywbf + (size_t)en * AHh + kk + cc * 8);
    }
#pragma unroll
    for (int i = 0; i < 4; ++i) {
      int idx = t + i * 256;
      int nc4 = idx & 15, kr = idx >> 4;
      float4 f = *(const float4*)(W + (size_t)(kk + kr) * Cc + n0 + nc4 * 4);
      sB[(nc4 * 4 + 0) * 64 + kr] = (__bf16)f.x;
      sB[(nc4 * 4 + 1) * 64 + kr] = (__bf16)f.y;
      sB[(nc4 * 4 + 2) * 64 + kr] = (__bf16)f.z;
      sB[(nc4 * 4 + 3) * 64 + kr] = (__bf16)f.w;
    }
    if (kk + 64 < AHh) __builtin_prefetch(W + (size_t)(kk + 64) * Cc + n0, 0, 1);
    wait_async0();
    __syncthreads();
#pragma unroll
    for (int ks = 0; ks < 64; ks += 32) {
      bf16x16 a  = lds_load_A(sA + msub * 16 * 64 + ks, 64, lane);
      bf16x16 b0 = lds_load_B(sB + (npair * 32) * 64 + ks, 64, lane);
      bf16x16 b1 = lds_load_B(sB + (npair * 32 + 16) * 64 + ks, 64, lane);
      acc0 = WMMA_BF16(a, b0, acc0);
      acc1 = WMMA_BF16(a, b1, acc1);
    }
    __syncthreads();
  }
  int n = lane & 15, mh = (lane & 16) ? 8 : 0;
#pragma unroll
  for (int r = 0; r < 8; ++r) {
    int row = msub * 16 + mh + r;
    int en  = srow[row];
    if (en >= 0) {
      int c0 = n0 + npair * 32 + n;
      yproj[(size_t)en * Cc + c0]      = acc0[r];
      yproj[(size_t)en * Cc + c0 + 16] = acc1[r];
    }
  }
}

// =====================================================================
// Kernel 6: combine two expert-slot contributions -> out (deterministic)
// =====================================================================
__global__ void combine_kernel(const float* __restrict__ yproj, float* __restrict__ out) {
  size_t i4 = (size_t)blockIdx.x * 256 + threadIdx.x;   // over BT*C/4
  size_t tok = i4 >> 8, c4 = i4 & 255;
  float4 a = *((const float4*)(yproj + (tok * 2) * Cc) + c4);
  float4 b = *((const float4*)(yproj + (tok * 2 + 1) * Cc) + c4);
  float4 r; r.x = a.x + b.x; r.y = a.y + b.y; r.z = a.z + b.z; r.w = a.w + b.w;
  *((float4*)(out + tok * Cc) + c4) = r;
}

__global__ void aux_kernel(const float* __restrict__ partials, const int* __restrict__ counts,
                           float* __restrict__ auxout) {
  if (threadIdx.x == 0) {
    float s[Ee];
#pragma unroll
    for (int e = 0; e < Ee; ++e) s[e] = 0.f;
    for (int blk = 0; blk < 128; ++blk)
#pragma unroll
      for (int e = 0; e < Ee; ++e) s[e] += partials[blk * Ee + e];
    float a = 0.f;
#pragma unroll
    for (int e = 0; e < Ee; ++e)
      a += (s[e] / (float)BT) * ((float)counts[e] / (float)BT);
    *auxout = (float)Ee * a;
  }
}

// =====================================================================
extern "C" void kernel_launch(void* const* d_in, const int* in_sizes, int n_in,
                              void* d_out, int out_size, void* d_ws, size_t ws_size,
                              hipStream_t stream) {
  (void)in_sizes; (void)n_in; (void)out_size; (void)ws_size;
  const float* x    = (const float*)d_in[0];
  const float* Wg   = (const float*)d_in[1];
  const float* W_in = (const float*)d_in[2];
  const float* W_out= (const float*)d_in[3];
  const float* Wk   = (const float*)d_in[4];
  const float* bk   = (const float*)d_in[5];
  const float* Wv   = (const float*)d_in[6];
  const float* bv   = (const float*)d_in[7];
  float* out = (float*)d_out;

  char* ws = (char*)d_ws;
  int*    counts   = (int*)   (ws + OFF_COUNTS);
  int*    lists    = (int*)   (ws + OFF_LISTS);
  float*  gates    = (float*) (ws + OFF_GATES);
  float*  partials = (float*) (ws + OFF_PART);
  __bf16* kbf      = (__bf16*)(ws + OFF_KBF);
  __bf16* vbf      = (__bf16*)(ws + OFF_VBF);
  __bf16* qbf      = (__bf16*)(ws + OFF_QBF);
  __bf16* ywbf     = (__bf16*)(ws + OFF_YWBF);
  float*  yproj    = (float*) (ws + OFF_YPROJ);

  hipMemsetAsync(counts, 0, 256, stream);

  gate_kernel<<<BT / 8, 256, 0, stream>>>(x, Wg, gates, counts, lists, partials);
  kv_kernel<<<dim3(AHh / 64, BT / 64, 2), 256, 0, stream>>>(x, Wk, bk, Wv, bv, kbf, vbf);
  qproj_kernel<<<dim3(AHh / 64, BT / 64, Ee), 256, 0, stream>>>(x, W_in, counts, lists, qbf);
  attn_kernel<<<dim3(Tt / 64, NHh, Bb * TOPK), 256, 0, stream>>>(qbf, kbf, vbf, gates, ywbf);
  oproj_kernel<<<dim3(Cc / 64, BT / 64, Ee), 256, 0, stream>>>(ywbf, W_out, counts, lists, yproj);
  combine_kernel<<<(BT * Cc / 4) / 256, 256, 0, stream>>>(yproj, out);
  aux_kernel<<<1, 32, 0, stream>>>(partials, counts, out + (size_t)BT * Cc);
}
